// RepeatNetModel_86818468921727
// MI455X (gfx1250) — matrix operand order repaired and checked by reading, not compiled, not run on hardware
//
#include <hip/hip_runtime.h>
#include <hip/hip_bf16.h>
#include <math.h>
#include <float.h>

typedef __bf16 bf16_t;
typedef __attribute__((ext_vector_type(16))) __bf16 bf16x16;
typedef __attribute__((ext_vector_type(8)))  __bf16 bf16x8;
typedef __attribute__((ext_vector_type(8)))  float  floatx8;

#define NEGBIG (-1e9f)

constexpr int Bz = 512, Sz = 50, Vz = 50000, Ez = 256, Hz = 256;

// ---------------------------------------------------------------------------
// block reductions
// ---------------------------------------------------------------------------
__device__ __forceinline__ float block_sum(float v, float* red) {
  int tid = threadIdx.x;
  red[tid] = v; __syncthreads();
  for (int o = 128; o > 0; o >>= 1) {
    if (tid < o) red[tid] += red[tid + o];
    __syncthreads();
  }
  float r = red[0]; __syncthreads();
  return r;
}

// ---------------------------------------------------------------------------
// fp32 -> bf16 conversion (grid-stride)
// ---------------------------------------------------------------------------
__global__ void cvt_bf16(const float* __restrict__ src, bf16_t* __restrict__ dst, int n) {
  int i = blockIdx.x * blockDim.x + threadIdx.x;
  int stride = gridDim.x * blockDim.x;
  for (; i < n; i += stride) dst[i] = (bf16_t)src[i];
}

// ---------------------------------------------------------------------------
// embedding gather -> bf16 x  (one block per (b,s) row)
// ---------------------------------------------------------------------------
__global__ __launch_bounds__(256) void gather_x(const int* __restrict__ seq,
                                                const float* __restrict__ emb,
                                                bf16_t* __restrict__ xb) {
  int row = blockIdx.x, tid = threadIdx.x;
  int it = seq[row];
  xb[(size_t)row * Ez + tid] = (bf16_t)emb[(size_t)it * Ez + tid];
}

// ---------------------------------------------------------------------------
// WMMA GEMM:  C[M,ldc] (fp32) = A[M,K] (bf16) * Bw[N,K]^T (bf16) + bias
// Block = 256 threads = 8 waves; block tile = 32(M) x 128(N).
// Each wave: one 16-col N-tile, TWO 16-row M-tiles sharing each B fragment
// (2 wmma per B load -> 32 FLOP per L2 byte, two independent acc chains).
// ---------------------------------------------------------------------------
__global__ __launch_bounds__(256) void gemm_bf16(const bf16_t* __restrict__ A,
                                                 const bf16_t* __restrict__ Bw,
                                                 float* __restrict__ C,
                                                 const float* __restrict__ bias,
                                                 int M, int N, int K, int ldc) {
  extern __shared__ bf16_t As[];
  const int lda_s = K + 8;
  int m0 = blockIdx.y << 5;  // 32 rows
  int tid = threadIdx.x;
  int Kv = K >> 3;  // 8-element chunks per row
  for (int i = tid; i < 32 * Kv; i += 256) {
    int r = i / Kv, kc = (i - r * Kv) << 3;
    *(bf16x8*)&As[r * lda_s + kc] = *(const bf16x8*)&A[(size_t)(m0 + r) * K + kc];
  }
  __syncthreads();
  int wave = tid >> 5, lane = tid & 31, half = lane >> 4, lrow = lane & 15;
  int nt = (blockIdx.x << 7) + (wave << 4);
  if (nt >= N) return;
  floatx8 acc0 = {0.f, 0.f, 0.f, 0.f, 0.f, 0.f, 0.f, 0.f};
  floatx8 acc1 = {0.f, 0.f, 0.f, 0.f, 0.f, 0.f, 0.f, 0.f};
  const bf16_t* brow = Bw + (size_t)(nt + lrow) * K;
  for (int kk = 0; kk < K; kk += 32) {
    // prefetch B stream ahead (global_prefetch_b8 path)
    __builtin_prefetch((const void*)(brow + kk + 256), 0, 0);
    // A-fragments: lanes 0-15 -> K = kk..kk+15; lanes 16-31 -> kk+16..kk+31 (contiguous)
    const bf16_t* ap0 = &As[lrow * lda_s + kk + half * 16];
    const bf16_t* ap1 = &As[(16 + lrow) * lda_s + kk + half * 16];
    bf16x8 a0lo = *(const bf16x8*)ap0;
    bf16x8 a0hi = *(const bf16x8*)(ap0 + 8);
    bf16x8 a1lo = *(const bf16x8*)ap1;
    bf16x8 a1hi = *(const bf16x8*)(ap1 + 8);
    bf16x16 a0, a1;
#pragma unroll
    for (int j = 0; j < 8; ++j) {
      a0[j] = a0lo[j]; a0[8 + j] = a0hi[j];
      a1[j] = a1lo[j]; a1[8 + j] = a1hi[j];
    }
    // B-fragment: lane column = nt+lrow, K = kk + half*16 .. +15 (contiguous)
    bf16x16 b = *(const bf16x16*)(brow + kk + half * 16);
    acc0 = __builtin_amdgcn_wmma_f32_16x16x32_bf16(false, a0, false, b, (short)0, acc0,
                                                   false, false);
    acc1 = __builtin_amdgcn_wmma_f32_16x16x32_bf16(false, a1, false, b, (short)0, acc1,
                                                   false, false);
  }
  int col = nt + lrow;
  float bv = bias ? bias[col] : 0.f;
  int mb = m0 + half * 8;
#pragma unroll
  for (int i = 0; i < 8; ++i)
    C[(size_t)(mb + i) * ldc + col] = acc0[i] + bv;
#pragma unroll
  for (int i = 0; i < 8; ++i)
    C[(size_t)(mb + 16 + i) * ldc + col] = acc1[i] + bv;
}

// ---------------------------------------------------------------------------
// GRU recurrence: 32 blocks x 16 batch rows, h kept in LDS, 50 sequential steps.
// Per step: gh[16,768] = h[16,256] @ Whh^T via WMMA, then fp32 gate math.
// ---------------------------------------------------------------------------
__global__ __launch_bounds__(256) void gru_kernel(const float* __restrict__ gi,
                                                  const bf16_t* __restrict__ Whh,
                                                  const float* __restrict__ bhh,
                                                  float* __restrict__ all_f,
                                                  bf16_t* __restrict__ all_b) {
  extern __shared__ float sm[];
  float* hf = sm;              // [16][260] fp32 hidden state
  float* gh = sm + 16 * 260;   // [16][772] fp32 gates
  int b0 = blockIdx.x << 4;
  int tid = threadIdx.x;
  int wave = tid >> 5, lane = tid & 31, half = lane >> 4, lrow = lane & 15;
  for (int i = tid; i < 16 * 256; i += 256) hf[(i >> 8) * 260 + (i & 255)] = 0.f;
  __syncthreads();
  for (int t = 0; t < Sz; ++t) {
    // gh = h @ Whh^T : 48 N-tiles over 8 waves = 6 tiles/wave
    for (int tt = 0; tt < 6; ++tt) {
      int nt = wave * 96 + tt * 16;
      floatx8 acc = {0.f, 0.f, 0.f, 0.f, 0.f, 0.f, 0.f, 0.f};
      const bf16_t* brow = Whh + (size_t)(nt + lrow) * 256;
#pragma unroll
      for (int kk = 0; kk < 256; kk += 32) {
        const float* ap = &hf[lrow * 260 + kk + half * 16];
        bf16x16 a;
#pragma unroll
        for (int j = 0; j < 8; ++j) {
          a[j]     = (bf16_t)ap[j];
          a[8 + j] = (bf16_t)ap[8 + j];
        }
        bf16x16 b = *(const bf16x16*)(brow + kk + half * 16);
        acc = __builtin_amdgcn_wmma_f32_16x16x32_bf16(false, a, false, b, (short)0, acc,
                                                      false, false);
      }
      int col = nt + lrow, mb = half * 8;
#pragma unroll
      for (int i = 0; i < 8; ++i) gh[(mb + i) * 772 + col] = acc[i];
    }
    __syncthreads();
    // elementwise gate math (PyTorch GRU cell)
    for (int idx = tid; idx < 16 * 256; idx += 256) {
      int r = idx >> 8, j = idx & 255;
      const float* gir = gi + ((size_t)(b0 + r) * Sz + t) * 768;
      float hr = gh[r * 772 + j]       + bhh[j];
      float hz = gh[r * 772 + 256 + j] + bhh[256 + j];
      float hn = gh[r * 772 + 512 + j] + bhh[512 + j];
      float rr = 1.f / (1.f + __expf(-(gir[j] + hr)));
      float zz = 1.f / (1.f + __expf(-(gir[256 + j] + hz)));
      float nn = tanhf(gir[512 + j] + rr * hn);
      float hnew = (1.f - zz) * nn + zz * hf[r * 260 + j];
      hf[r * 260 + j] = hnew;
      size_t o = ((size_t)(b0 + r) * Sz + t) * 256 + j;
      all_f[o] = hnew;
      all_b[o] = (bf16_t)hnew;
    }
    __syncthreads();
  }
}

// ---------------------------------------------------------------------------
// last hidden state gather (fp32 + bf16)
// ---------------------------------------------------------------------------
__global__ __launch_bounds__(256) void gather_last(const float* __restrict__ all_f,
                                                   const int* __restrict__ lens,
                                                   float* __restrict__ last_f,
                                                   bf16_t* __restrict__ last_b) {
  int b = blockIdx.x, j = threadIdx.x;
  int t = lens[b] - 1;
  float v = all_f[((size_t)b * Sz + t) * Hz + j];
  last_f[b * Hz + j] = v;
  last_b[b * Hz + j] = (bf16_t)v;
}

// ---------------------------------------------------------------------------
// attention scores: one block per (b,s) row; three tanh-dot-reductions
// ---------------------------------------------------------------------------
__global__ __launch_bounds__(256) void scores_kernel(
    const float* __restrict__ P1, const float* __restrict__ P2,
    const float* __restrict__ P3, const float* __restrict__ q1,
    const float* __restrict__ q2, const float* __restrict__ q3,
    const float* __restrict__ Vre, const float* __restrict__ Vr,
    const float* __restrict__ Ve, const float* __restrict__ Vr_b,
    const float* __restrict__ Ve_b, const int* __restrict__ seq,
    float* __restrict__ s_re, float* __restrict__ s_r, float* __restrict__ s_e) {
  __shared__ float red[256];
  int i = blockIdx.x;  // row in [0, B*S)
  int b = i / Sz;
  int tid = threadIdx.x;
  size_t ro = (size_t)i * Hz + tid;
  int qo = b * Hz + tid;
  float v1 = tanhf(P1[ro] + q1[qo]) * Vre[tid];
  float v2 = tanhf(P2[ro] + q2[qo]) * Vr[tid];
  float v3 = tanhf(P3[ro] + q3[qo]) * Ve[tid];
  float r1 = block_sum(v1, red);
  float r2 = block_sum(v2, red);
  float r3 = block_sum(v3, red);
  if (tid == 0) {
    bool pad = (seq[i] == 0);
    s_re[i] = r1;
    s_r[i]  = pad ? NEGBIG : (r2 + Vr_b[0]);
    s_e[i]  = pad ? NEGBIG : (r3 + Ve_b[0]);
  }
}

// ---------------------------------------------------------------------------
// softmaxes over S, context sums, repeat/explore gate, out_e assembly
// ---------------------------------------------------------------------------
__global__ __launch_bounds__(256) void sm_ctx_kernel(
    const float* __restrict__ s_re, const float* __restrict__ s_r,
    const float* __restrict__ s_e, const float* __restrict__ all_f,
    const float* __restrict__ last_f, const float* __restrict__ Wcre,
    float* __restrict__ p_r, float* __restrict__ rep_exp,
    bf16_t* __restrict__ out_e) {
  __shared__ float are[Sz], ae[Sz];
  __shared__ float red[256];
  int b = blockIdx.x, tid = threadIdx.x;
  if (tid == 0) {  // softmax(s_re) -> are
    float mx = -1e30f;
    for (int s = 0; s < Sz; ++s) mx = fmaxf(mx, s_re[b * Sz + s]);
    float smv = 0.f;
    for (int s = 0; s < Sz; ++s) { float e = __expf(s_re[b * Sz + s] - mx); are[s] = e; smv += e; }
    float r = 1.f / smv;
    for (int s = 0; s < Sz; ++s) are[s] *= r;
  } else if (tid == 32) {  // masked softmax(s_r) -> p_r (global)
    float mx = -1e30f;
    for (int s = 0; s < Sz; ++s) mx = fmaxf(mx, s_r[b * Sz + s]);
    float smv = 0.f;
    for (int s = 0; s < Sz; ++s) smv += __expf(s_r[b * Sz + s] - mx);
    float r = 1.f / smv;
    for (int s = 0; s < Sz; ++s) p_r[b * Sz + s] = __expf(s_r[b * Sz + s] - mx) * r;
  } else if (tid == 64) {  // masked softmax(s_e) -> ae
    float mx = -1e30f;
    for (int s = 0; s < Sz; ++s) mx = fmaxf(mx, s_e[b * Sz + s]);
    float smv = 0.f;
    for (int s = 0; s < Sz; ++s) { float e = __expf(s_e[b * Sz + s] - mx); ae[s] = e; smv += e; }
    float r = 1.f / smv;
    for (int s = 0; s < Sz; ++s) ae[s] *= r;
  }
  __syncthreads();
  float c1 = 0.f, c2 = 0.f;
  for (int s = 0; s < Sz; ++s) {
    float m = all_f[((size_t)b * Sz + s) * Hz + tid];
    c1 += are[s] * m;
    c2 += ae[s] * m;
  }
  out_e[(size_t)b * (2 * Hz) + tid]      = (bf16_t)c2;
  out_e[(size_t)b * (2 * Hz) + Hz + tid] = (bf16_t)last_f[b * Hz + tid];
  float d0 = block_sum(c1 * Wcre[tid], red);
  float d1 = block_sum(c1 * Wcre[Hz + tid], red);
  if (tid == 0) {
    float mx = fmaxf(d0, d1);
    float e0 = __expf(d0 - mx), e1 = __expf(d1 - mx);
    float inv = 1.f / (e0 + e1);
    rep_exp[b * 2]     = e0 * inv;
    rep_exp[b * 2 + 1] = e1 * inv;
  }
}

// ---------------------------------------------------------------------------
// in-place over d_out logits: mask seen items, ONLINE softmax over V (single
// max+sum pass, float4 vectorized), scale by gate, atomic-add repeat scatter.
// One block per batch row.
// ---------------------------------------------------------------------------
__global__ __launch_bounds__(256) void final_kernel(float* __restrict__ out,
                                                    const int* __restrict__ seq,
                                                    const float* __restrict__ p_r,
                                                    const float* __restrict__ rep_exp) {
  __shared__ int items[Sz];
  __shared__ float prs[Sz];
  __shared__ float rm[256], rs[256];
  int b = blockIdx.x, tid = threadIdx.x;
  if (tid < Sz) { items[tid] = seq[b * Sz + tid]; prs[tid] = p_r[b * Sz + tid]; }
  __syncthreads();
  float* row = out + (size_t)b * Vz;
  if (tid < Sz && items[tid] > 0) row[items[tid]] = -INFINITY;
  __threadfence();
  __syncthreads();
  // online (max, sum) in one pass; seed max with -FLT_MAX so (-inf) entries
  // never produce inf-inf = NaN in the rescale term.
  float m = -FLT_MAX, s = 0.f;
  const float4* row4 = (const float4*)row;
  for (int i = tid; i < Vz / 4; i += 256) {
    float4 x4 = row4[i];
    float xm = fmaxf(fmaxf(x4.x, x4.y), fmaxf(x4.z, x4.w));
    if (xm > m) { s *= __expf(m - xm); m = xm; }
    s += __expf(x4.x - m) + __expf(x4.y - m) + __expf(x4.z - m) + __expf(x4.w - m);
  }
  // combine (m, s) pairs across the block
  rm[tid] = m; rs[tid] = s; __syncthreads();
  for (int o = 128; o > 0; o >>= 1) {
    if (tid < o) {
      float m2 = rm[tid + o], s2 = rs[tid + o];
      float M = fmaxf(rm[tid], m2);
      rs[tid] = rs[tid] * __expf(rm[tid] - M) + s2 * __expf(m2 - M);
      rm[tid] = M;
    }
    __syncthreads();
  }
  float M = rm[0], S = rs[0];
  __syncthreads();
  float wr = rep_exp[b * 2], we = rep_exp[b * 2 + 1];
  float inv = we / S;
  float4* row4w = (float4*)row;
  for (int i = tid; i < Vz / 4; i += 256) {
    float4 x4 = row4w[i];
    x4.x = __expf(x4.x - M) * inv;
    x4.y = __expf(x4.y - M) * inv;
    x4.z = __expf(x4.z - M) * inv;
    x4.w = __expf(x4.w - M) * inv;
    row4w[i] = x4;
  }
  __threadfence();
  __syncthreads();
  if (tid < Sz && items[tid] > 0) atomicAdd(&row[items[tid]], prs[tid] * wr);
}

// ---------------------------------------------------------------------------
// host orchestration
// ---------------------------------------------------------------------------
extern "C" void kernel_launch(void* const* d_in, const int* in_sizes, int n_in,
                              void* d_out, int out_size, void* d_ws, size_t ws_size,
                              hipStream_t stream) {
  const int*   item_seq = (const int*)d_in[0];
  const int*   seq_len  = (const int*)d_in[1];
  const float* emb      = (const float*)d_in[2];
  const float* Wih      = (const float*)d_in[3];
  const float* Whh      = (const float*)d_in[4];
  const float* bih      = (const float*)d_in[5];
  const float* bhh      = (const float*)d_in[6];
  const float* Wre      = (const float*)d_in[7];
  const float* Ure      = (const float*)d_in[8];
  const float* Vre      = (const float*)d_in[9];
  const float* Wcre     = (const float*)d_in[10];
  const float* Wr       = (const float*)d_in[11];
  const float* Ur       = (const float*)d_in[12];
  const float* Vr_w     = (const float*)d_in[13];
  const float* Vr_b     = (const float*)d_in[14];
  const float* We_w     = (const float*)d_in[15];
  const float* We_b     = (const float*)d_in[16];
  const float* Ue_w     = (const float*)d_in[17];
  const float* Ue_b     = (const float*)d_in[18];
  const float* Ve_w     = (const float*)d_in[19];
  const float* Ve_b     = (const float*)d_in[20];
  const float* Wexp     = (const float*)d_in[21];

  char* ws = (char*)d_ws;
  size_t off = 0;
  auto take = [&](size_t bytes) -> char* {
    char* p = ws + off;
    off += (bytes + 255) & ~(size_t)255;
    return p;
  };
  bf16_t* xb     = (bf16_t*)take((size_t)Bz * Sz * Ez * 2);   // bf16 x
  bf16_t* Wih_b  = (bf16_t*)take((size_t)768 * 256 * 2);
  bf16_t* Whh_b  = (bf16_t*)take((size_t)768 * 256 * 2);
  bf16_t* Ure_b  = (bf16_t*)take((size_t)256 * 256 * 2);
  bf16_t* Wre_b  = (bf16_t*)take((size_t)256 * 256 * 2);
  bf16_t* Ur_b   = (bf16_t*)take((size_t)256 * 256 * 2);
  bf16_t* Wr_b   = (bf16_t*)take((size_t)256 * 256 * 2);
  bf16_t* Ue_b2  = (bf16_t*)take((size_t)256 * 256 * 2);
  bf16_t* We_b2  = (bf16_t*)take((size_t)256 * 256 * 2);
  bf16_t* Wexp_b = (bf16_t*)take((size_t)Vz * 512 * 2);
  float*  gi     = (float*)take((size_t)Bz * Sz * 768 * 4);   // reused as P1/P2/P3 later
  float*  all_f  = (float*)take((size_t)Bz * Sz * Hz * 4);
  bf16_t* all_b  = (bf16_t*)take((size_t)Bz * Sz * Hz * 2);
  float*  last_f = (float*)take((size_t)Bz * Hz * 4);
  bf16_t* last_b = (bf16_t*)take((size_t)Bz * Hz * 2);
  float*  q1     = (float*)take((size_t)Bz * Hz * 4);
  float*  q2     = (float*)take((size_t)Bz * Hz * 4);
  float*  q3     = (float*)take((size_t)Bz * Hz * 4);
  float*  s_re   = (float*)take((size_t)Bz * Sz * 4);
  float*  s_r    = (float*)take((size_t)Bz * Sz * 4);
  float*  s_e    = (float*)take((size_t)Bz * Sz * 4);
  float*  p_r    = (float*)take((size_t)Bz * Sz * 4);
  float*  repexp = (float*)take((size_t)Bz * 2 * 4);
  bf16_t* out_e  = (bf16_t*)take((size_t)Bz * 512 * 2);
  // alias: projection buffers reuse gi (dead after GRU)
  float* P1 = gi;
  float* P2 = gi + (size_t)Bz * Sz * Hz;
  float* P3 = gi + (size_t)2 * Bz * Sz * Hz;

  float* out = (float*)d_out;

  // 1) weight conversions
  cvt_bf16<<<1024, 256, 0, stream>>>(Wih,  Wih_b,  768 * 256);
  cvt_bf16<<<1024, 256, 0, stream>>>(Whh,  Whh_b,  768 * 256);
  cvt_bf16<<<1024, 256, 0, stream>>>(Ure,  Ure_b,  256 * 256);
  cvt_bf16<<<1024, 256, 0, stream>>>(Wre,  Wre_b,  256 * 256);
  cvt_bf16<<<1024, 256, 0, stream>>>(Ur,   Ur_b,   256 * 256);
  cvt_bf16<<<1024, 256, 0, stream>>>(Wr,   Wr_b,   256 * 256);
  cvt_bf16<<<1024, 256, 0, stream>>>(Ue_w, Ue_b2,  256 * 256);
  cvt_bf16<<<1024, 256, 0, stream>>>(We_w, We_b2,  256 * 256);
  cvt_bf16<<<4096, 256, 0, stream>>>(Wexp, Wexp_b, Vz * 512);

  // 2) embedding gather
  gather_x<<<Bz * Sz, 256, 0, stream>>>(item_seq, emb, xb);

  // 3) gi = x @ Wih^T + bih   [25600 x 768], K=256
  {
    size_t lds = (size_t)32 * (256 + 8) * 2;
    gemm_bf16<<<dim3(6, Bz * Sz / 32), 256, lds, stream>>>(xb, Wih_b, gi, bih,
                                                           Bz * Sz, 768, 256, 768);
  }

  // 4) GRU recurrence
  {
    size_t lds = (size_t)16 * 260 * 4 + (size_t)16 * 772 * 4;
    gru_kernel<<<Bz / 16, 256, lds, stream>>>(gi, Whh_b, bhh, all_f, all_b);
  }

  // 5) last hidden state
  gather_last<<<Bz, 256, 0, stream>>>(all_f, seq_len, last_f, last_b);

  // 6) attention projections (all_mem: M=25600; last: M=512; N=K=256)
  {
    size_t lds = (size_t)32 * (256 + 8) * 2;
    gemm_bf16<<<dim3(2, Bz * Sz / 32), 256, lds, stream>>>(all_b, Ure_b, P1, nullptr,
                                                           Bz * Sz, 256, 256, 256);
    gemm_bf16<<<dim3(2, Bz * Sz / 32), 256, lds, stream>>>(all_b, Ur_b, P2, nullptr,
                                                           Bz * Sz, 256, 256, 256);
    gemm_bf16<<<dim3(2, Bz * Sz / 32), 256, lds, stream>>>(all_b, Ue_b2, P3, Ue_b,
                                                           Bz * Sz, 256, 256, 256);
    gemm_bf16<<<dim3(2, Bz / 32), 256, lds, stream>>>(last_b, Wre_b, q1, nullptr,
                                                      Bz, 256, 256, 256);
    gemm_bf16<<<dim3(2, Bz / 32), 256, lds, stream>>>(last_b, Wr_b, q2, nullptr,
                                                      Bz, 256, 256, 256);
    gemm_bf16<<<dim3(2, Bz / 32), 256, lds, stream>>>(last_b, We_b2, q3, We_b,
                                                      Bz, 256, 256, 256);
  }

  // 7) attention scores
  scores_kernel<<<Bz * Sz, 256, 0, stream>>>(P1, P2, P3, q1, q2, q3, Vre, Vr_w, Ve_w,
                                             Vr_b, Ve_b, item_seq, s_re, s_r, s_e);

  // 8) softmaxes over S, contexts, gate, out_e
  sm_ctx_kernel<<<Bz, 256, 0, stream>>>(s_re, s_r, s_e, all_f, last_f, Wcre, p_r,
                                        repexp, out_e);

  // 9) explore logits straight into d_out: [512 x 50000], K=512
  {
    size_t lds = (size_t)32 * (512 + 8) * 2;
    gemm_bf16<<<dim3((Vz + 127) / 128, Bz / 32), 256, lds, stream>>>(
        out_e, Wexp_b, out, nullptr, Bz, Vz, 512, Vz);
  }

  // 10) mask + online softmax over V + gate mix + repeat scatter (in place)
  final_kernel<<<Bz, 256, 0, stream>>>(out, item_seq, p_r, repexp);
}
